// CosSim2D_10050223472950
// MI455X (gfx1250) — matrix operand order repaired and checked by reading, not compiled, not run on hardware
//
#include <hip/hip_runtime.h>
#include <hip/hip_bf16.h>
#include <math.h>

typedef __attribute__((ext_vector_type(16))) _Float16 v16h;
typedef __attribute__((ext_vector_type(8)))  _Float16 v8h;
typedef __attribute__((ext_vector_type(8)))  float    v8f;

#define C_IN   64
#define N_K    128
#define KTOT   576
#define HDIM   64
#define WDIM   64
#define ROWPAD 72   // 64 channels + 8 halves pad -> 144B row stride, conflict-free b128 reads

union AFrag { v16h v; v8h h[2]; };

// ---------------------------------------------------------------------------
// Prep: w (576,128) f32 -> wT (128,576) f16, columns L2-normalized.
// One block per output kernel n; 64 threads reduce the 576-term square sum.
// ---------------------------------------------------------------------------
__global__ __launch_bounds__(64) void prep_w(const float* __restrict__ w,
                                             _Float16* __restrict__ wT)
{
    __shared__ float red[64];
    const int n = blockIdx.x;
    const int t = threadIdx.x;
    float vals[9];
    float s = 0.f;
#pragma unroll
    for (int i = 0; i < 9; ++i) {
        const int k = t + i * 64;
        const float v = w[k * N_K + n];
        vals[i] = v;
        s += v * v;
    }
    red[t] = s;
    __syncthreads();
#pragma unroll
    for (int off = 32; off > 0; off >>= 1) {
        if (t < off) red[t] += red[t + off];
        __syncthreads();
    }
    const float scale = 1.0f / sqrtf(fmaxf(red[0], 1e-12f));
#pragma unroll
    for (int i = 0; i < 9; ++i) {
        const int k = t + i * 64;
        wT[n * KTOT + k] = (_Float16)(vals[i] * scale);
    }
}

// ---------------------------------------------------------------------------
// Main: implicit-im2col cosine-sim GEMM.
// Block = one (b,h) row: 64 pixels x 128 kernels. 256 threads = 8 waves,
// tiled 4(M) x 2(N); each wave owns four 16x16 f32 accumulators.
// ---------------------------------------------------------------------------
__global__ __launch_bounds__(256) void cossim_main(const float* __restrict__ x,
                                                   const _Float16* __restrict__ wT,
                                                   const float* __restrict__ p,
                                                   const float* __restrict__ q,
                                                   float* __restrict__ out)
{
    __shared__ _Float16 Abuf[2][64 * ROWPAD];
    __shared__ float normpart[256];
    __shared__ float xnorm_rcp[64];

    const int tid = threadIdx.x;
    const int blk = blockIdx.x;
    const int b   = blk >> 6;
    const int h   = blk & 63;

    // staging role: 4 threads per pixel, 16 channels each
    const int pix = tid >> 2;            // 0..63  == w coordinate
    const int c0  = (tid & 3) << 4;      // 0,16,32,48

    const float q_eff = __expf(-q[0] * (1.0f / 0.3f));

    float normacc = 0.f;

    auto load_tap = [&](int t, float4* rr) {
        const int dy = t / 3 - 1;
        const int dx = t % 3 - 1;
        const int hh = h + dy;
        const int ww = pix + dx;
        if (hh >= 0 && hh < HDIM && ww >= 0 && ww < WDIM) {
            const float4* src =
                (const float4*)(x + ((size_t)((b * HDIM + hh) * WDIM + ww) * C_IN + c0));
            rr[0] = src[0]; rr[1] = src[1]; rr[2] = src[2]; rr[3] = src[3];
        } else {
            const float4 z = make_float4(0.f, 0.f, 0.f, 0.f);
            rr[0] = z; rr[1] = z; rr[2] = z; rr[3] = z;
        }
    };

    auto commit = [&](int buf, const float4* rr) {
        _Float16 hv[16];
        float s = 0.f;
#pragma unroll
        for (int u = 0; u < 4; ++u) {
            const float4 f = rr[u];
            s += f.x * f.x + f.y * f.y + f.z * f.z + f.w * f.w;
            hv[u * 4 + 0] = (_Float16)f.x;
            hv[u * 4 + 1] = (_Float16)f.y;
            hv[u * 4 + 2] = (_Float16)f.z;
            hv[u * 4 + 3] = (_Float16)f.w;
        }
        normacc += s;
        v8h* dst = (v8h*)&Abuf[buf][pix * ROWPAD + c0];
        dst[0] = *(const v8h*)&hv[0];
        dst[1] = *(const v8h*)&hv[8];
    };

    // prologue: stage tap 0
    {
        float4 r0[4];
        load_tap(0, r0);
        commit(0, r0);
    }
    __syncthreads();

    // compute role
    const int wave  = tid >> 5;
    const int lane  = tid & 31;
    const int mw    = wave >> 1;         // 0..3 : 16-pixel strip
    const int nw    = wave & 1;          // 0..1 : 64-kernel half
    const int mloc  = mw * 16 + (lane & 15);
    const int lhalf = lane >> 4;         // 0/1

    v8f acc0 = {}, acc1 = {}, acc2 = {}, acc3 = {};

    for (int t = 0; t < 9; ++t) {
        float4 rn[4];
        if (t < 8) {
            load_tap(t + 1, rn);                    // global loads overlap WMMA below
            if (t < 7) {                            // warm caches two taps ahead
                const int dy = (t + 2) / 3 - 1, dx = (t + 2) % 3 - 1;
                const int hh = h + dy, ww = pix + dx;
                if (hh >= 0 && hh < HDIM && ww >= 0 && ww < WDIM)
                    __builtin_prefetch(
                        x + ((size_t)((b * HDIM + hh) * WDIM + ww) * C_IN + c0), 0, 1);
            }
        }

        const _Float16* Ab = &Abuf[t & 1][0];
#pragma unroll
        for (int kb = 0; kb < 64; kb += 32) {
            // A fragment: 16 consecutive K per lane half (two 8-half chunks at k and k+8)
            AFrag a;
            const _Float16* abase = Ab + mloc * ROWPAD + kb + (lhalf << 4);
            a.h[0] = *(const v8h*)(abase);
            a.h[1] = *(const v8h*)(abase + 8);

            // B fragments: lane = column n, 16 consecutive K from N-major wT
            const int kg = t * 64 + kb + (lhalf << 4);
            const _Float16* Bb = wT + kg + (size_t)(nw * 64 + (lane & 15)) * KTOT;

            v16h b0 = *(const v16h*)(Bb + 0 * 16 * KTOT);
            acc0 = __builtin_amdgcn_wmma_f32_16x16x32_f16(false, a.v, false, b0,
                                                          (short)0, acc0, false, false);
            v16h b1 = *(const v16h*)(Bb + 1 * 16 * KTOT);
            acc1 = __builtin_amdgcn_wmma_f32_16x16x32_f16(false, a.v, false, b1,
                                                          (short)0, acc1, false, false);
            v16h b2 = *(const v16h*)(Bb + 2 * 16 * KTOT);
            acc2 = __builtin_amdgcn_wmma_f32_16x16x32_f16(false, a.v, false, b2,
                                                          (short)0, acc2, false, false);
            v16h b3 = *(const v16h*)(Bb + 3 * 16 * KTOT);
            acc3 = __builtin_amdgcn_wmma_f32_16x16x32_f16(false, a.v, false, b3,
                                                          (short)0, acc3, false, false);
        }

        if (t < 8) {
            commit((t + 1) & 1, rn);     // write the buffer not being read this tap
            __syncthreads();
        }
    }

    // per-pixel L2 norm reduction (4 partials per pixel); store RECIPROCAL once
    normpart[tid] = normacc;
    __syncthreads();
    if (tid < 64) {
        const float s = normpart[tid * 4 + 0] + normpart[tid * 4 + 1] +
                        normpart[tid * 4 + 2] + normpart[tid * 4 + 3];
        xnorm_rcp[tid] = 1.0f / (sqrtf(fmaxf(s, 1e-12f)) + q_eff);
    }
    __syncthreads();

    // epilogue: sim * rcp_norm, sign(x)*(|x|+eps)^p_eff via exp(p*log(.)),
    // non-temporal fp32 stores (pure 67MB output stream, keep caches for x/wT)
    const int m0 = blk * 64;
    const v8f accs[4] = {acc0, acc1, acc2, acc3};
#pragma unroll
    for (int j = 0; j < 4; ++j) {
        const int n  = nw * 64 + j * 16 + (lane & 15);
        const float pe = __expf(p[n] * 0.2f);       // exp(p / P_SCALE)
#pragma unroll
        for (int rr = 0; rr < 8; ++rr) {
            const int ml  = mw * 16 + rr + (lhalf << 3);   // D layout: M = r + 8*(lane>=16)
            const float sim = accs[j][rr] * xnorm_rcp[ml];
            // base = |sim|+eps is strictly positive & finite: pow == exp(pe*log(base))
            const float base = fabsf(sim) + 1e-6f;
            const float v    = __expf(pe * __logf(base));
            __builtin_nontemporal_store(copysignf(v, sim),
                                        &out[(size_t)(m0 + ml) * N_K + n]);
        }
    }
}

// ---------------------------------------------------------------------------
extern "C" void kernel_launch(void* const* d_in, const int* in_sizes, int n_in,
                              void* d_out, int out_size, void* d_ws, size_t ws_size,
                              hipStream_t stream) {
    const float* x = (const float*)d_in[0];   // (32,64,64,64) f32
    const float* w = (const float*)d_in[1];   // (1,576,128) f32
    const float* p = (const float*)d_in[2];   // (128,) f32
    const float* q = (const float*)d_in[3];   // (1,) f32
    float* out = (float*)d_out;               // (32,64,64,128) f32
    _Float16* wT = (_Float16*)d_ws;           // 128*576 f16 = 147456 B

    prep_w<<<N_K, 64, 0, stream>>>(w, wT);
    cossim_main<<<32 * HDIM, 256, 0, stream>>>(x, wT, p, q, out);
}
